// MoEViTBlock_11132555231274
// MI455X (gfx1250) — matrix operand order repaired and checked by reading, not compiled, not run on hardware
//
#include <hip/hip_runtime.h>
#include <hip/hip_bf16.h>
#include <math.h>

// ---------------- problem constants ----------------
constexpr int C_   = 768;
constexpr int E_   = 8;
constexpr int H_   = 3072;
constexpr int TOK  = 16384;      // 16 * 1024
constexpr int CAP  = 5120;       // int(1.25 * TOK * 2 / 8)

typedef __attribute__((ext_vector_type(16))) __bf16 v16bf;
typedef __attribute__((ext_vector_type(8)))  float  v8f;
typedef __attribute__((ext_vector_type(4)))  int    v4i;

union FragAB {
    float4 f[2];
    v16bf  v;
};
static_assert(sizeof(FragAB) == 32, "frag size");

// ---------------- CDNA5 async global->LDS copy support ----------------
#if defined(__has_builtin)
#if __has_builtin(__builtin_amdgcn_global_load_async_to_lds_b128)
#define USE_ASYNC_COPY 1
#endif
#endif
#ifndef USE_ASYNC_COPY
#define USE_ASYNC_COPY 0
#endif

// builtin signature: (int4 addrspace(1)* gaddr, int4 addrspace(3)* ldsaddr, imm offset, imm cpol)
typedef __attribute__((address_space(1))) v4i* gbl_v4i_p;
typedef __attribute__((address_space(3))) v4i* lds_v4i_p;
#define AS_GLOBAL(p) ((gbl_v4i_p)(p))
#define AS_LDS(p)    ((lds_v4i_p)(p))

__device__ __forceinline__ void wait_asynccnt_zero() {
#if defined(__has_builtin) && __has_builtin(__builtin_amdgcn_s_wait_asynccnt)
    __builtin_amdgcn_s_wait_asynccnt(0);
#else
    asm volatile("s_wait_asynccnt 0x0" ::: "memory");
#endif
}

// ---------------- workspace layout (bytes) ----------------
constexpr size_t S_W1T  = (size_t)E_ * H_ * C_;        // bf16 elements, [E][H][C] (transposed W1)
constexpr size_t S_W2T  = (size_t)E_ * C_ * H_;        // bf16 elements, [E][C][H] (transposed W2)
constexpr size_t S_BUFX = (size_t)E_ * CAP * C_;       // bf16
constexpr size_t S_H    = (size_t)E_ * CAP * H_;       // bf16
constexpr size_t S_EOUT = (size_t)E_ * CAP * C_;       // f32

constexpr size_t O_W1T   = 0;
constexpr size_t O_W2T   = O_W1T  + S_W1T  * 2;
constexpr size_t O_BUFX  = O_W2T  + S_W2T  * 2;
constexpr size_t O_H     = O_BUFX + S_BUFX * 2;
constexpr size_t O_EOUT  = O_H    + S_H    * 2;
constexpr size_t O_ASSGN = O_EOUT + S_EOUT * 4;
constexpr size_t O_TOKW  = O_ASSGN + (size_t)2 * TOK * 4;
constexpr size_t O_CNT   = O_TOKW  + (size_t)2 * TOK * 4;

// ---------------- 32x32 tiled transpose fp32 -> bf16, per expert ----------------
__global__ __launch_bounds__(256) void transpose_to_bf16(
    const float* __restrict__ W, __bf16* __restrict__ Wt, int K, int N)
{
    __shared__ float tile[32][33];
    const int e  = blockIdx.z;
    const float* Wp  = W  + (size_t)e * K * N;
    __bf16*      Wtp = Wt + (size_t)e * K * N;
    const int n0 = blockIdx.x * 32;
    const int k0 = blockIdx.y * 32;
    const int tx = threadIdx.x & 31;
    const int ty = threadIdx.x >> 5;       // 0..7
#pragma unroll
    for (int p = 0; p < 4; ++p) {
        int k = ty + p * 8;
        tile[k][tx] = Wp[(size_t)(k0 + k) * N + n0 + tx];   // coalesced read
    }
    __syncthreads();
#pragma unroll
    for (int p = 0; p < 4; ++p) {
        int n = ty + p * 8;
        Wtp[(size_t)(n0 + n) * K + k0 + tx] = (__bf16)tile[tx][n];  // coalesced write
    }
}

// ---------------- router: logits, top-2 softmax, capacity slots ----------------
__global__ __launch_bounds__(256) void router_kernel(
    const float* __restrict__ xf, const float* __restrict__ Wg,
    float* __restrict__ tokW, int* __restrict__ assign_row, int* __restrict__ cnt)
{
    const int lane = threadIdx.x & 31;
    const int t    = blockIdx.x * 8 + (threadIdx.x >> 5);   // one wave per token
    const float* xr = xf + (size_t)t * C_;
    float acc[E_];
#pragma unroll
    for (int e = 0; e < E_; ++e) acc[e] = 0.f;
    for (int c = lane; c < C_; c += 32) {
        float xv = xr[c];
        const float* wg = Wg + c * E_;
#pragma unroll
        for (int e = 0; e < E_; ++e) acc[e] += xv * wg[e];
    }
#pragma unroll
    for (int e = 0; e < E_; ++e) {
#pragma unroll
        for (int off = 16; off > 0; off >>= 1)
            acc[e] += __shfl_xor(acc[e], off, 32);
    }
    if (lane == 0) {
        int e0 = 0;
#pragma unroll
        for (int e = 1; e < E_; ++e) if (acc[e] > acc[e0]) e0 = e;
        int e1 = (e0 == 0) ? 1 : 0;
#pragma unroll
        for (int e = 0; e < E_; ++e) if (e != e0 && acc[e] > acc[e1]) e1 = e;
        float v0 = acc[e0], v1 = acc[e1];
        float w0 = 1.f / (1.f + expf(v1 - v0));   // softmax over top-2
        float w1 = 1.f - w0;
        int p0 = atomicAdd(&cnt[e0], 1);
        int p1 = atomicAdd(&cnt[e1], 1);
        assign_row[2 * t + 0] = (p0 < CAP) ? (e0 * CAP + p0) : -1;
        assign_row[2 * t + 1] = (p1 < CAP) ? (e1 * CAP + p1) : -1;
        tokW[2 * t + 0] = w0;
        tokW[2 * t + 1] = w1;
    }
}

// ---------------- gather tokens into expert buffers (fp32 -> bf16) ----------------
__global__ __launch_bounds__(256) void gather_kernel(
    const float* __restrict__ xf, const int* __restrict__ assign_row,
    __bf16* __restrict__ bufx)
{
    const int a   = blockIdx.x;
    const int row = assign_row[a];
    if (row < 0) return;
    const int t = a >> 1;
    const float* src = xf   + (size_t)t   * C_;
    __bf16*      dst = bufx + (size_t)row * C_;
    for (int i = threadIdx.x; i < C_; i += 256)
        dst[i] = (__bf16)src[i];
}

// ---------------- tiled WMMA bf16 GEMM: Out = act(A[M,K] * Bt[N,K]^T + bias) ----------------
// A row-major [M][K]; Bt is the weight pre-transposed to [N][K]; per-expert batch in blockIdx.z.
// Block tile 256(M) x 128(N), 8 waves in a 4x2 grid, 64x64 per wave (4x4 WMMA frags).
template<bool GELU_BF16_OUT>
__global__ __launch_bounds__(256) void moe_gemm(
    const __bf16* __restrict__ A, const __bf16* __restrict__ Bt,
    const float* __restrict__ bias, void* __restrict__ Out,
    int M, int N, int K)
{
    constexpr int KS    = 32;  // K per step (bf16 wmma depth)
    constexpr int LDS_K = 40;  // padded K stride in LDS (bank-conflict avoidance)
    constexpr int BM    = 256;
    constexpr int BN    = 128;

    __shared__ __bf16 Alds[2][BM * LDS_K];   // 2 x 20 KB
    __shared__ __bf16 Blds[2][BN * LDS_K];   // 2 x 10 KB

    const int e  = blockIdx.z;
    const __bf16* Ae = A  + (size_t)e * M * K;
    const __bf16* Be = Bt + (size_t)e * N * K;
    const float*  be = bias + (size_t)e * N;

    const int tm = blockIdx.y * BM;
    const int tn = blockIdx.x * BN;

    const int tid  = threadIdx.x;
    const int lane = tid & 31;
    const int wid  = tid >> 5;
    const int wm   = wid & 3;        // 4 waves along M (64 rows each)
    const int wn   = wid >> 2;       // 2 waves along N (64 cols each)
    const int lr   = lane & 15;
    const int hi   = lane >> 4;      // lane half

    const int KT = K / KS;

    // staging: A = 256x32 (4 x b128/thread), B = 128x32 (2 x b128/thread)
    const int srow_q = tid & 3;
    const int srow_0 = tid >> 2;

    auto stage = [&](int kt, int b) {
        const int kk = kt * KS;
#pragma unroll
        for (int p = 0; p < 4; ++p) {
            const int row = srow_0 + 64 * p;
            const __bf16* ga = Ae + (size_t)(tm + row) * K + kk + srow_q * 8;
            __bf16*       la = &Alds[b][row * LDS_K + srow_q * 8];
#if USE_ASYNC_COPY
            __builtin_amdgcn_global_load_async_to_lds_b128(AS_GLOBAL(ga), AS_LDS(la), 0, 0);
#else
            *reinterpret_cast<float4*>(la) = *reinterpret_cast<const float4*>(ga);
#endif
        }
#pragma unroll
        for (int p = 0; p < 2; ++p) {
            const int row = srow_0 + 64 * p;
            const __bf16* gb = Be + (size_t)(tn + row) * K + kk + srow_q * 8;
            __bf16*       lb = &Blds[b][row * LDS_K + srow_q * 8];
#if USE_ASYNC_COPY
            __builtin_amdgcn_global_load_async_to_lds_b128(AS_GLOBAL(gb), AS_LDS(lb), 0, 0);
#else
            *reinterpret_cast<float4*>(lb) = *reinterpret_cast<const float4*>(gb);
#endif
        }
        // prefetch the next step into GL2 (global_prefetch_b8)
        if (kt + 1 < KT) {
            __builtin_prefetch(Ae + (size_t)(tm + srow_0) * K + kk + KS, 0, 1);
            __builtin_prefetch(Be + (size_t)(tn + srow_0) * K + kk + KS, 0, 1);
        }
    };

    v8f acc[4][4];
#pragma unroll
    for (int i = 0; i < 4; ++i)
#pragma unroll
        for (int j = 0; j < 4; ++j)
#pragma unroll
            for (int v = 0; v < 8; ++v) acc[i][j][v] = 0.f;

    stage(0, 0);

    for (int kt = 0; kt < KT; ++kt) {
        const int cb = kt & 1;

        // make this wave's async copies land, then make all waves' copies visible
#if USE_ASYNC_COPY
        wait_asynccnt_zero();
#endif
        __syncthreads();

        // kick off the next tile's copies into the other buffer (overlaps the WMMAs below)
        if (kt + 1 < KT) stage(kt + 1, cb ^ 1);

        FragAB a[4], b[4];
#pragma unroll
        for (int i = 0; i < 4; ++i) {
            const int r = wm * 64 + i * 16 + lr;
            const __bf16* base = &Alds[cb][r * LDS_K + hi * 8];
            a[i].f[0] = *reinterpret_cast<const float4*>(base);        // K chunk [b, b+8)
            a[i].f[1] = *reinterpret_cast<const float4*>(base + 16);   // K chunk [b+16, b+24)
        }
#pragma unroll
        for (int j = 0; j < 4; ++j) {
            const int c = wn * 64 + j * 16 + lr;
            const __bf16* base = &Blds[cb][c * LDS_K + hi * 8];
            b[j].f[0] = *reinterpret_cast<const float4*>(base);        // K chunk [b, b+8)
            b[j].f[1] = *reinterpret_cast<const float4*>(base + 16);   // K chunk [b+16, b+24)
        }
#pragma unroll
        for (int i = 0; i < 4; ++i)
#pragma unroll
            for (int j = 0; j < 4; ++j)
                acc[i][j] = __builtin_amdgcn_wmma_f32_16x16x32_bf16(
                    false, a[i].v, false, b[j].v, (short)0, acc[i][j], false, false);
    }

    // epilogue: D layout -> m = base_m + 8*hi + v, n = base_n + lr
#pragma unroll
    for (int j = 0; j < 4; ++j) {
        const int n  = tn + wn * 64 + j * 16 + lr;
        const float bv = be[n];
#pragma unroll
        for (int i = 0; i < 4; ++i) {
            const int mbase = tm + wm * 64 + i * 16 + hi * 8;
#pragma unroll
            for (int v = 0; v < 8; ++v) {
                const int m = mbase + v;
                float val = acc[i][j][v] + bv;
                if (GELU_BF16_OUT) {
                    val = 0.5f * val * (1.f + erff(val * 0.70710678118654752f));
                    ((__bf16*)Out)[(size_t)e * M * N + (size_t)m * N + n] = (__bf16)val;
                } else {
                    ((float*)Out)[(size_t)e * M * N + (size_t)m * N + n] = val;
                }
            }
        }
    }
}

// ---------------- weighted scatter back to tokens ----------------
__global__ __launch_bounds__(256) void scatter_kernel(
    const float* __restrict__ eout, const int* __restrict__ assign_row,
    const float* __restrict__ tokW, float* __restrict__ y)
{
    const int t  = blockIdx.x;
    const int r0 = assign_row[2 * t + 0];
    const int r1 = assign_row[2 * t + 1];
    const float w0 = tokW[2 * t + 0];
    const float w1 = tokW[2 * t + 1];
    float* yr = y + (size_t)t * C_;
    for (int i = threadIdx.x; i < C_; i += 256) {
        float s = 0.f;
        if (r0 >= 0) s += w0 * eout[(size_t)r0 * C_ + i];
        if (r1 >= 0) s += w1 * eout[(size_t)r1 * C_ + i];
        yr[i] = s;
    }
}

// ---------------- launcher ----------------
extern "C" void kernel_launch(void* const* d_in, const int* in_sizes, int n_in,
                              void* d_out, int out_size, void* d_ws, size_t ws_size,
                              hipStream_t stream) {
    (void)in_sizes; (void)n_in; (void)out_size; (void)ws_size;

    const float* x  = (const float*)d_in[0];   // [16384, 768]
    const float* Wg = (const float*)d_in[1];   // [768, 8]
    const float* W1 = (const float*)d_in[2];   // [8, 768, 3072]
    const float* b1 = (const float*)d_in[3];   // [8, 3072]
    const float* W2 = (const float*)d_in[4];   // [8, 3072, 768]
    const float* b2 = (const float*)d_in[5];   // [8, 768]
    float* y = (float*)d_out;

    char* ws = (char*)d_ws;
    __bf16* w1t  = (__bf16*)(ws + O_W1T);
    __bf16* w2t  = (__bf16*)(ws + O_W2T);
    __bf16* bufx = (__bf16*)(ws + O_BUFX);
    __bf16* hbuf = (__bf16*)(ws + O_H);
    float*  eout = (float*) (ws + O_EOUT);
    int*    asgn = (int*)   (ws + O_ASSGN);
    float*  tokw = (float*) (ws + O_TOKW);
    int*    cnt  = (int*)   (ws + O_CNT);

    (void)hipMemsetAsync(cnt, 0, E_ * sizeof(int), stream);

    // weights fp32 -> bf16, transposed to [N][K]
    transpose_to_bf16<<<dim3(H_ / 32, C_ / 32, E_), 256, 0, stream>>>(W1, w1t, C_, H_);
    transpose_to_bf16<<<dim3(C_ / 32, H_ / 32, E_), 256, 0, stream>>>(W2, w2t, H_, C_);

    // routing + capacity assignment
    router_kernel<<<TOK / 8, 256, 0, stream>>>(x, Wg, tokw, asgn, cnt);

    // gather tokens into expert buffers (bf16)
    gather_kernel<<<2 * TOK, 256, 0, stream>>>(x, asgn, bufx);

    // expert MLP: h = gelu(bufx @ W1 + b1)   (M=5120, N=3072, K=768)
    moe_gemm<true><<<dim3(H_ / 128, CAP / 256, E_), 256, 0, stream>>>(
        bufx, w1t, b1, (void*)hbuf, CAP, H_, C_);

    // expert MLP: eout = h @ W2 + b2         (M=5120, N=768, K=3072)
    moe_gemm<false><<<dim3(C_ / 128, CAP / 256, E_), 256, 0, stream>>>(
        hbuf, w2t, b2, (void*)eout, CAP, C_, H_);

    // weighted scatter back to token order
    scatter_kernel<<<TOK, 256, 0, stream>>>(eout, asgn, tokw, y);
}
